// SFMCNN_1769526526666
// MI455X (gfx1250) — compile-verified, matrix operand order checked by reading, not run on hardware
//
#include <hip/hip_runtime.h>
#include <hip/hip_bf16.h>
#include <cstddef>
#include <cstdint>

// ---------------------------------------------------------------------------
// CDNA5 (gfx1250) SFMCNN pipeline.
// rbf_conv == GEMM (cross term, f16 WMMA w/ f32 accum) + fp32 norm vectors.
// gauss(d)=exp(-d^2/(2s^2)) consumes d^2 directly -> no sqrt anywhere.
// K-dimension of conv GEMMs is permuted to (ky,kx,c) so NHWC im2col staging
// is contiguous; weights are pre-permuted identically (dot is order-invariant).
// B tiles are moved global->LDS by the Tensor Data Mover when available.
// ---------------------------------------------------------------------------

typedef __attribute__((ext_vector_type(16))) _Float16 v16h;
typedef __attribute__((ext_vector_type(8)))  _Float16 v8h;
typedef __attribute__((ext_vector_type(4)))  _Float16 v4h;
typedef __attribute__((ext_vector_type(8)))  float    v8f;
typedef __attribute__((ext_vector_type(4)))  float    v4f;
typedef __attribute__((ext_vector_type(4)))  unsigned int v4u;
typedef __attribute__((ext_vector_type(8)))  int      v8i;
typedef __attribute__((ext_vector_type(4)))  int      v4i;

union U16h { v16h v; v8h h[2]; v4h q[4]; _Float16 e[16]; };

static __device__ __forceinline__ v8f wmma_f32_f16(v16h a, v16h b, v8f c) {
  return __builtin_amdgcn_wmma_f32_16x16x32_f16(
      /*neg_a=*/false, a, /*neg_b=*/false, b,
      /*c_mod=*/(short)0, c, /*reuse_a=*/false, /*reuse_b=*/false);
}

#if defined(__gfx1250__) && __has_builtin(__builtin_amdgcn_tensor_load_to_lds) && \
    __has_builtin(__builtin_amdgcn_s_wait_tensorcnt)
#define SFM_USE_TDM 1
#endif

#ifdef SFM_USE_TDM
// TDM: move a [rows x 32 halves] tile (row stride = strideElems halves) from
// global into LDS at ldsOff, LDS-padding 4 DWORDs after every 16 DWORDs so the
// LDS row pitch is 40 halves (matches LDB). Rows >= 'rows' are zero-filled.
static __device__ __forceinline__ void tdm_load_b_tile(const _Float16* gsrc,
                                                       unsigned ldsOff,
                                                       int rows,
                                                       int strideElems) {
  union { v4u v; unsigned d[4]; } g0;
  union { v8i v; int d[8]; } g1;
  unsigned long long ga = (unsigned long long)(uintptr_t)gsrc;
  g0.d[0] = 1u;                                   // count=1 (valid), user mode
  g0.d[1] = ldsOff;                               // lds_addr (bytes)
  g0.d[2] = (unsigned)ga;                         // global_addr[31:0]
  g0.d[3] = (unsigned)((ga >> 32) & 0x1FFFFFFu)   // global_addr[56:32]
            | (2u << 30);                         // type = 2 ("image")
  // data_size=2B, pad_enable, pad_interval=16 DWORDs, pad_amount=4 DWORDs
  g1.d[0] = (int)((1u << 16) | (1u << 20) | (3u << 22) | (3u << 25));
  g1.d[1] = (int)(32u << 16);                     // tensor_dim0 = 32
  g1.d[2] = (int)((unsigned)rows << 16);          // tensor_dim1 = rows
  g1.d[3] = (int)(32u << 16);                     // tile_dim0 = 32
  g1.d[4] = 64;                                   // tile_dim1 = 64
  g1.d[5] = strideElems;                          // tensor_dim0_stride
  g1.d[6] = 0;
  g1.d[7] = 0;
  v4i gz = {0, 0, 0, 0};
#if __clang_major__ >= 23
  v8i gx = {0, 0, 0, 0, 0, 0, 0, 0};
  __builtin_amdgcn_tensor_load_to_lds(g0.v, g1.v, gz, gz, gx, 0);
#else
  __builtin_amdgcn_tensor_load_to_lds(g0.v, g1.v, gz, gz, 0);
#endif
}
#endif

// ---------------------------------------------------------------------------
// Weight prep: f32 [N][Kreal] -> f16 [N][Kpad] (zero padded) + wn[N] = |w|^2.
// perm=1 reorders K from (c,ky,kx) to (ky,kx,c).
// ---------------------------------------------------------------------------
__global__ __launch_bounds__(256) void prep_w(const float* __restrict__ w,
                                              _Float16* __restrict__ wq,
                                              float* __restrict__ wn,
                                              int Kreal, int Kpad,
                                              int C, int kk2, int perm) {
  const int row = blockIdx.x;
  const int t   = threadIdx.x;
  const float* wr = w + (size_t)row * Kreal;
  _Float16*    qr = wq + (size_t)row * Kpad;
  float s = 0.f;
  for (int kk = t; kk < Kpad; kk += 256) {
    float v = 0.f;
    if (kk < Kreal) {
      int src = kk;
      if (perm) {
        int sp = kk / C;
        int c  = kk - sp * C;
        src = c * kk2 + sp;
      }
      v = wr[src];
    }
    qr[kk] = (_Float16)v;
    s += v * v;
  }
  __shared__ float red[256];
  red[t] = s;
  __syncthreads();
  for (int o = 128; o > 0; o >>= 1) {
    if (t < o) red[t] += red[t + o];
    __syncthreads();
  }
  if (t == 0) wn[row] = red[0];
}

// ---------------------------------------------------------------------------
// Per-pixel channel squared sum: s[b,y,x] = sum_c in(b,c,y,x)^2
// ---------------------------------------------------------------------------
__global__ __launch_bounds__(256) void sq_chan(const float* __restrict__ in,
                                               float* __restrict__ s,
                                               long long sb, long long sc,
                                               long long sy, long long sx,
                                               int C, int wShift, int hShift,
                                               int total) {
  int t = blockIdx.x * blockDim.x + threadIdx.x;
  if (t >= total) return;
  const int x = t & ((1 << wShift) - 1);
  const int y = (t >> wShift) & ((1 << hShift) - 1);
  const int b = t >> (wShift + hShift);
  const float* p = in + (long long)b * sb + (long long)y * sy + (long long)x * sx;
  float acc = 0.f;
  for (int c = 0; c < C; ++c) {
    float v = p[(long long)c * sc];
    acc += v * v;
  }
  s[t] = acc;
}

// ---------------------------------------------------------------------------
// Window sum of s over kxk window (stride 1, zero pad): pn[b,oy,ox]
// ---------------------------------------------------------------------------
__global__ __launch_bounds__(256) void pn_win(const float* __restrict__ s,
                                              float* __restrict__ pn,
                                              int k, int pad,
                                              int wShift, int hShift, int total) {
  int t = blockIdx.x * blockDim.x + threadIdx.x;
  if (t >= total) return;
  const int W = 1 << wShift, H = 1 << hShift;
  const int x = t & (W - 1);
  const int y = (t >> wShift) & (H - 1);
  const int b = t >> (wShift + hShift);
  const float* sb_ = s + (size_t)b * H * W;
  float acc = 0.f;
  for (int ky = 0; ky < k; ++ky) {
    int iy = y + ky - pad;
    if ((unsigned)iy >= (unsigned)H) continue;
    for (int kx = 0; kx < k; ++kx) {
      int ix = x + kx - pad;
      if ((unsigned)ix >= (unsigned)W) continue;
      acc += sb_[iy * W + ix];
    }
  }
  pn[t] = acc;
}

// ---------------------------------------------------------------------------
// RBF-conv GEMM.  Tile: 128(M) x 64(N), Kc=32.  8 waves: 4x2 grid of 32x32
// wave tiles, each wave = 2x2 V_WMMA_F32_16X16X32_F16.
// PERM=0: K order (c,ky,kx), scalar clamped gathers (layer 1, NCHW, C=3).
// PERM=1: K order (ky,kx,c), contiguous b128 gathers (NHWC, C%16==0).
// OUTMODE 0: h as f16 NHWC; OUTMODE 1: h as f32 NCHW.
// ---------------------------------------------------------------------------
constexpr int BM = 128, BN = 64, KC = 32, LDA = 40, LDB = 40;

template <int OUTMODE, int PERM>
__global__ __launch_bounds__(256) void rbf_gemm(
    const float* __restrict__ in,
    long long sb, long long sc, long long sy, long long sx,
    int C, int H, int W, int k, int pad,
    int woShift, int hoShift,
    const _Float16* __restrict__ wq, int Kpad,
    const float* __restrict__ pn, const float* __restrict__ wnrm,
    const float* __restrict__ stdp, const float* __restrict__ biasp,
    int Ntot, void* __restrict__ outp) {
  __shared__ _Float16 As[BM * LDA];
  __shared__ _Float16 Bs[BN * LDB];

  const int tid  = threadIdx.x;
  const int lane = tid & 31;
  const int wid  = tid >> 5;
  const int wm   = wid & 3;
  const int wn2  = wid >> 2;
  const int blockM = blockIdx.x * BM;
  const int blockN = blockIdx.y * BN;
  const int Wo  = 1 << woShift;
  const int kk2 = k * k;

  // A loader: thread -> (row, 16-col half) of the 128x32 tile
  const int arow  = tid >> 1;
  const int acol0 = (tid & 1) * 16;
  const int m   = blockM + arow;
  const int ox  = m & (Wo - 1);
  const int oy  = (m >> woShift) & ((1 << hoShift) - 1);
  const int bb  = m >> (woShift + hoShift);
  const long long inb = (long long)bb * sb;

  // B tile geometry
  const int nvalid = (Ntot - blockN) < BN ? (Ntot - blockN) : BN;
#ifndef SFM_USE_TDM
  const int brow  = tid & 63;
  const int bcol0 = (tid >> 6) * 8;
  const int bn = blockN + brow;
  const bool bvalid = bn < Ntot;
  const _Float16* brp = wq + (size_t)bn * Kpad;
#endif

  v8f acc[2][2];
  {
    v8f z = {0.f, 0.f, 0.f, 0.f, 0.f, 0.f, 0.f, 0.f};
    acc[0][0] = z; acc[0][1] = z; acc[1][0] = z; acc[1][1] = z;
  }

  for (int kb = 0; kb < Kpad; kb += KC) {
    // ---- stage A tile: on-the-fly im2col, f32 -> f16 ----
    if (PERM == 1) {
      // K order (ky,kx,c): a 16-run is contiguous channels at one (ky,kx).
      int kkstart = kb + acol0;
      int sp = kkstart / C;
      int c0 = kkstart - sp * C;
      int ky = sp / k;
      int kx = sp - ky * k;
      int iy = oy + ky - pad;
      int ix = ox + kx - pad;
      v8h* adst = (v8h*)&As[arow * LDA + acol0];
      if (((unsigned)iy < (unsigned)H) & ((unsigned)ix < (unsigned)W)) {
        const float* src = in + inb + (long long)iy * sy + (long long)ix * sx + c0;
        U16h tmp;
        tmp.q[0] = __builtin_convertvector(*(const v4f*)&src[0],  v4h);
        tmp.q[1] = __builtin_convertvector(*(const v4f*)&src[4],  v4h);
        tmp.q[2] = __builtin_convertvector(*(const v4f*)&src[8],  v4h);
        tmp.q[3] = __builtin_convertvector(*(const v4f*)&src[12], v4h);
        adst[0] = tmp.h[0];
        adst[1] = tmp.h[1];
      } else {
        v8h zz = { (_Float16)0, (_Float16)0, (_Float16)0, (_Float16)0,
                   (_Float16)0, (_Float16)0, (_Float16)0, (_Float16)0 };
        adst[0] = zz;
        adst[1] = zz;
      }
    } else {
      // K order (c,ky,kx): scalar gathers with clamped addresses (no exec
      // branching around loads), select-to-zero afterwards.
      int kkstart = kb + acol0;
      int c  = kkstart / kk2;
      int r  = kkstart - c * kk2;
      int ky = r / k;
      int kx = r - ky * k;
      _Float16* adst = &As[arow * LDA + acol0];
      for (int j = 0; j < 16; ++j) {
        int iy = oy + ky - pad;
        int ix = ox + kx - pad;
        bool valid = (c < C) & ((unsigned)iy < (unsigned)H) &
                     ((unsigned)ix < (unsigned)W);
        int cc  = c  >= C ? C - 1 : c;
        int iyc = iy < 0 ? 0 : (iy >= H ? H - 1 : iy);
        int ixc = ix < 0 ? 0 : (ix >= W ? W - 1 : ix);
        float v = in[inb + (long long)cc * sc + (long long)iyc * sy +
                     (long long)ixc * sx];
        adst[j] = (_Float16)(valid ? v : 0.f);
        if (++kx == k) { kx = 0; if (++ky == k) { ky = 0; ++c; } }
      }
    }

    // ---- stage B tile ----
#ifdef SFM_USE_TDM
    if (wid == 0) {
      tdm_load_b_tile(wq + (size_t)blockN * Kpad + kb,
                      (unsigned)(uintptr_t)&Bs[0], nvalid, Kpad);
      __builtin_amdgcn_s_wait_tensorcnt(0);
    }
#else
    {
      v8h bv = { (_Float16)0, (_Float16)0, (_Float16)0, (_Float16)0,
                 (_Float16)0, (_Float16)0, (_Float16)0, (_Float16)0 };
      if (bvalid) {
        bv = *(const v8h*)&brp[kb + bcol0];
        if (kb + KC < Kpad) __builtin_prefetch(&brp[kb + KC + bcol0], 0, 0);
      }
      *(v8h*)&Bs[brow * LDB + bcol0] = bv;
    }
#endif
    __syncthreads();

    // ---- fragments (documented 16-bit WMMA VGPR layout) + 4x WMMA ----
    {
      const int khOff = (lane >> 4) * 8;
      const int lc    = lane & 15;
      U16h a0, a1, b0, b1;
      a0.h[0] = *(const v8h*)&As[(wm * 32 +      lc) * LDA +      khOff];
      a0.h[1] = *(const v8h*)&As[(wm * 32 +      lc) * LDA + 16 + khOff];
      a1.h[0] = *(const v8h*)&As[(wm * 32 + 16 + lc) * LDA +      khOff];
      a1.h[1] = *(const v8h*)&As[(wm * 32 + 16 + lc) * LDA + 16 + khOff];
      b0.h[0] = *(const v8h*)&Bs[(wn2 * 32 +      lc) * LDB +      khOff];
      b0.h[1] = *(const v8h*)&Bs[(wn2 * 32 +      lc) * LDB + 16 + khOff];
      b1.h[0] = *(const v8h*)&Bs[(wn2 * 32 + 16 + lc) * LDB +      khOff];
      b1.h[1] = *(const v8h*)&Bs[(wn2 * 32 + 16 + lc) * LDB + 16 + khOff];
      acc[0][0] = wmma_f32_f16(a0.v, b0.v, acc[0][0]);
      acc[0][1] = wmma_f32_f16(a0.v, b1.v, acc[0][1]);
      acc[1][0] = wmma_f32_f16(a1.v, b0.v, acc[1][0]);
      acc[1][1] = wmma_f32_f16(a1.v, b1.v, acc[1][1]);
    }
    __syncthreads();
  }

  // ---- epilogue: d2 -> gauss -> crelu -> store ----
  const float stdv  = stdp[0];
  const float biasv = biasp[0];
  const float inv2  = 1.0f / (2.0f * stdv * stdv);
  const int lc  = lane & 15;
  const int lhi = (lane >> 4) * 8;
  for (int mi = 0; mi < 2; ++mi) {
    const int mgBase = blockM + wm * 32 + mi * 16 + lhi;
    const v4f pn0 = *(const v4f*)&pn[mgBase];
    const v4f pn1 = *(const v4f*)&pn[mgBase + 4];
    for (int ni = 0; ni < 2; ++ni) {
      int ng = blockN + wn2 * 32 + ni * 16 + lc;
      if (ng >= Ntot) continue;
      float wnv = wnrm[ng];
#pragma unroll
      for (int rr = 0; rr < 8; ++rr) {
        int mg = mgBase + rr;
        float pnm = (rr < 4) ? pn0[rr] : pn1[rr - 4];
        float d2 = pnm + wnv - 2.0f * acc[mi][ni][rr];
        d2 = fmaxf(d2, 0.f);
        float g = __expf(-d2 * inv2);
        float h = (g >= biasv) ? g : 0.f;
        if (OUTMODE == 0) {
          ((_Float16*)outp)[(size_t)mg * Ntot + ng] = (_Float16)h;
        } else {
          const int hw  = woShift + hoShift;
          const int b2  = mg >> hw;
          const int rem = mg & ((1 << hw) - 1);
          ((float*)outp)[(((size_t)b2 * Ntot + ng) << hw) + rem] = h;
        }
      }
    }
  }
}

// ---------------------------------------------------------------------------
// SFM: 2x2 non-overlapping alpha-weighted mean pool, NHWC f16 -> NHWC f32.
// ---------------------------------------------------------------------------
__global__ void sfm_pool(const _Float16* __restrict__ h,
                         float* __restrict__ out,
                         const float* __restrict__ alpha,
                         int C, int inW, int inH, int owShift, int ohShift) {
  const int pix = blockIdx.x;
  const int c   = threadIdx.x;
  const int oW  = 1 << owShift;
  const int ox  = pix & (oW - 1);
  const int oy  = (pix >> owShift) & ((1 << ohShift) - 1);
  const int b   = pix >> (owShift + ohShift);
  const int iy = oy * 2, ix = ox * 2;
  const size_t rowStride = (size_t)inW * C;
  const size_t base = (((size_t)b * inH + iy) * inW + ix) * C + c;
  float v = alpha[0] * (float)h[base] +
            alpha[1] * (float)h[base + C] +
            alpha[2] * (float)h[base + rowStride] +
            alpha[3] * (float)h[base + rowStride + C];
  out[(size_t)pix * C + c] = 0.25f * v;
}

// ---------------------------------------------------------------------------
// FC: out = h3[16,K] @ fc_w[100,K]^T + b.  Deterministic split-K WMMA:
// grid (7 n-tiles, 256 k-slices), one wave per workgroup, branch-free b128
// loads (invalid columns read a clamped row; their results are never reduced).
// ---------------------------------------------------------------------------
__global__ __launch_bounds__(32) void fc_gemv(const float* __restrict__ h3,
                                              const float* __restrict__ fcw,
                                              float* __restrict__ part,
                                              int K, int Ncols) {
  const int nt = blockIdx.x;   // 0..6
  const int s  = blockIdx.y;   // 0..255
  const int l  = threadIdx.x;
  const int Kw = K >> 8;       // 1536
  const int k0 = s * Kw;
  const int mrow = l & 15;
  const int kh   = (l >> 4) * 8;
  int col = nt * 16 + (l & 15);
  if (col >= Ncols) col = Ncols - 1;   // clamp; unused columns never reduced
  const float* arow = h3 + (size_t)mrow * K;
  const float* brow = fcw + (size_t)col * K;

  v8f acc = {0.f, 0.f, 0.f, 0.f, 0.f, 0.f, 0.f, 0.f};
  for (int kc = k0; kc < k0 + Kw; kc += 32) {
    U16h a, b;
    a.q[0] = __builtin_convertvector(*(const v4f*)&arow[kc + kh],      v4h);
    a.q[1] = __builtin_convertvector(*(const v4f*)&arow[kc + kh + 4],  v4h);
    a.q[2] = __builtin_convertvector(*(const v4f*)&arow[kc + 16 + kh], v4h);
    a.q[3] = __builtin_convertvector(*(const v4f*)&arow[kc + 20 + kh], v4h);
    b.q[0] = __builtin_convertvector(*(const v4f*)&brow[kc + kh],      v4h);
    b.q[1] = __builtin_convertvector(*(const v4f*)&brow[kc + kh + 4],  v4h);
    b.q[2] = __builtin_convertvector(*(const v4f*)&brow[kc + 16 + kh], v4h);
    b.q[3] = __builtin_convertvector(*(const v4f*)&brow[kc + 20 + kh], v4h);
    acc = wmma_f32_f16(a.v, b.v, acc);
  }
  float* pbase = part + (size_t)(nt * 256 + s) * 256;
#pragma unroll
  for (int r = 0; r < 8; ++r) {
    int mm = (l >> 4) * 8 + r;
    int nn = l & 15;
    pbase[mm * 16 + nn] = acc[r];
  }
}

__global__ __launch_bounds__(256) void fc_reduce(const float* __restrict__ part,
                                                 const float* __restrict__ fcb,
                                                 float* __restrict__ out) {
  int t = blockIdx.x * blockDim.x + threadIdx.x;
  if (t >= 16 * 100) return;
  const int b  = t / 100;
  const int n  = t - b * 100;
  const int nt = n >> 4;
  const int nn = n & 15;
  float sum = fcb[n];
  const float* p = part + (size_t)nt * 256 * 256 + b * 16 + nn;
  for (int s = 0; s < 256; ++s) sum += p[(size_t)s * 256];
  out[t] = sum;
}

// ---------------------------------------------------------------------------
extern "C" void kernel_launch(void* const* d_in, const int* in_sizes, int n_in,
                              void* d_out, int out_size, void* d_ws, size_t ws_size,
                              hipStream_t stream) {
  (void)in_sizes; (void)n_in; (void)out_size; (void)ws_size;
  const float* x      = (const float*)d_in[0];
  const float* w1     = (const float*)d_in[1];
  const float* std1   = (const float*)d_in[2];
  const float* bias1  = (const float*)d_in[3];
  const float* alpha1 = (const float*)d_in[4];
  const float* w2     = (const float*)d_in[5];
  const float* std2   = (const float*)d_in[6];
  const float* bias2  = (const float*)d_in[7];
  const float* alpha2 = (const float*)d_in[8];
  const float* w3     = (const float*)d_in[9];
  const float* std3   = (const float*)d_in[10];
  const float* bias3  = (const float*)d_in[11];
  const float* fcw    = (const float*)d_in[12];
  const float* fcb    = (const float*)d_in[13];
  float* out = (float*)d_out;

  // ---- workspace carve-out (256B aligned) ----
  char* ws = (char*)d_ws;
  size_t off = 0;
  auto alloc = [&](size_t bytes) -> char* {
    char* p = ws + off;
    off = (off + bytes + 255) & ~(size_t)255;
    return p;
  };
  _Float16* w1q = (_Float16*)alloc((size_t)96 * 96 * 2);
  float*    wn1 = (float*)alloc(96 * 4);
  _Float16* w2q = (_Float16*)alloc((size_t)192 * 2400 * 2);
  float*    wn2 = (float*)alloc(192 * 4);
  _Float16* w3q = (_Float16*)alloc((size_t)384 * 1728 * 2);
  float*    wn3 = (float*)alloc(384 * 4);
  float*    s1  = (float*)alloc((size_t)262144 * 4);
  float*    pn1 = (float*)alloc((size_t)262144 * 4);
  _Float16* h1  = (_Float16*)alloc((size_t)262144 * 96 * 2);  // NHWC f16 full-res
  float*    a1  = (float*)alloc((size_t)65536 * 96 * 4);      // NHWC f32 pooled
  float*    s2  = (float*)alloc((size_t)65536 * 4);
  float*    pn2 = (float*)alloc((size_t)65536 * 4);
  _Float16* h2  = (_Float16*)alloc((size_t)65536 * 192 * 2);
  float*    a2  = (float*)alloc((size_t)16384 * 192 * 4);
  float*    s3  = (float*)alloc((size_t)16384 * 4);
  float*    pn3 = (float*)alloc((size_t)16384 * 4);
  float*    h3  = (float*)alloc((size_t)16384 * 384 * 4);     // NCHW f32 (FC order)
  float*    fcp = (float*)alloc((size_t)7 * 256 * 256 * 4);

  // ---- weight prep (layer1 identity K-order; layers 2/3 permuted) ----
  prep_w<<<96,  256, 0, stream>>>(w1, w1q, wn1, 75,   96,   3,   25, 0);
  prep_w<<<192, 256, 0, stream>>>(w2, w2q, wn2, 2400, 2400, 96,  25, 1);
  prep_w<<<384, 256, 0, stream>>>(w3, w3q, wn3, 1728, 1728, 192, 9,  1);

  // ---- layer 1: x NCHW [16,3,128,128], k=5 pad=2 -> h1 f16 NHWC [16,128,128,96]
  sq_chan<<<1024, 256, 0, stream>>>(x, s1, 49152LL, 16384LL, 128LL, 1LL, 3, 7, 7, 262144);
  pn_win<<<1024, 256, 0, stream>>>(s1, pn1, 5, 2, 7, 7, 262144);
  rbf_gemm<0, 0><<<dim3(2048, 2), 256, 0, stream>>>(
      x, 49152LL, 16384LL, 128LL, 1LL, 3, 128, 128, 5, 2, 7, 7,
      w1q, 96, pn1, wn1, std1, bias1, 96, (void*)h1);
  sfm_pool<<<65536, 96, 0, stream>>>(h1, a1, alpha1, 96, 128, 128, 6, 6);

  // ---- layer 2: a1 NHWC [16,64,64,96], k=5 pad=2 -> h2 f16 NHWC [16,64,64,192]
  sq_chan<<<256, 256, 0, stream>>>(a1, s2, 393216LL, 1LL, 6144LL, 96LL, 96, 6, 6, 65536);
  pn_win<<<256, 256, 0, stream>>>(s2, pn2, 5, 2, 6, 6, 65536);
  rbf_gemm<0, 1><<<dim3(512, 3), 256, 0, stream>>>(
      a1, 393216LL, 1LL, 6144LL, 96LL, 96, 64, 64, 5, 2, 6, 6,
      w2q, 2400, pn2, wn2, std2, bias2, 192, (void*)h2);
  sfm_pool<<<16384, 192, 0, stream>>>(h2, a2, alpha2, 192, 64, 64, 5, 5);

  // ---- layer 3: a2 NHWC [16,32,32,192], k=3 pad=1 -> h3 f32 NCHW [16,384,32,32]
  sq_chan<<<64, 256, 0, stream>>>(a2, s3, 196608LL, 1LL, 6144LL, 192LL, 192, 5, 5, 16384);
  pn_win<<<64, 256, 0, stream>>>(s3, pn3, 3, 1, 5, 5, 16384);
  rbf_gemm<1, 1><<<dim3(128, 6), 256, 0, stream>>>(
      a2, 196608LL, 1LL, 6144LL, 192LL, 192, 32, 32, 3, 1, 5, 5,
      w3q, 1728, pn3, wn3, std3, bias3, 384, (void*)h3);

  // ---- FC: [16, 393216] @ [100, 393216]^T + b -> [16,100]
  fc_gemv<<<dim3(7, 256), 32, 0, stream>>>(h3, fcw, fcp, 393216, 100);
  fc_reduce<<<7, 256, 0, stream>>>(fcp, fcb, out);
}
